// SyncGrokMoeBlock_1726576856636
// MI455X (gfx1250) — matrix operand order repaired and checked by reading, not compiled, not run on hardware
//
#include <hip/hip_runtime.h>
#include <hip/hip_bf16.h>

// ---------------- CDNA5 WMMA types ----------------
typedef __attribute__((ext_vector_type(16))) __bf16 v16bf;
typedef __attribute__((ext_vector_type(8)))  float  v8f;

struct __align__(16) Q4 { unsigned int a, b, c, d; };
union Frag { Q4 q[2]; v16bf v; };

// ---------------- Problem constants ----------------
constexpr int kT = 8192;     // B*S tokens
constexpr int kH = 1024;     // hidden
constexpr int kF = 4096;     // ffn
constexpr int kE = 8;        // experts
constexpr int MT = 64;       // M tile (slots)
constexpr int NT = 64;       // N tile
constexpr int KT = 32;       // K tile (bf16 wmma K)
constexpr int KTP = 40;      // padded LDS row stride (ushorts): 80B, 16B aligned, conflict-light
constexpr int kSlotMax = kT * 2 + kE * MT;   // 16896
constexpr int kBlkM = kSlotMax / MT;         // 264

// ---------------- Workspace layout (bytes) ----------------
constexpr size_t offCtrl = 0;                                   // 32 ints: counts[8], cursors[8], offsets[9]
constexpr size_t offSelE = 256;
constexpr size_t offSelW = offSelE + (size_t)kT * 2 * 4;
constexpr size_t offTok  = offSelW + (size_t)kT * 2 * 4;
constexpr size_t offSwt  = offTok + (size_t)kSlotMax * 4;
constexpr size_t offXbf  = (offSwt + (size_t)kSlotMax * 4 + 255) & ~(size_t)255; // x bf16 [T][H]
constexpr size_t offHws  = offXbf + (size_t)kT * kH * 2;        // h bf16 [kSlotMax][F]
constexpr size_t offWgT  = offHws + (size_t)kSlotMax * kF * 2;  // wg bf16 transposed [E][F][H]
constexpr size_t offWvT  = offWgT + (size_t)kE * kH * kF * 2;   // wv bf16 transposed [E][F][H]
constexpr size_t offW1T  = offWvT + (size_t)kE * kH * kF * 2;   // w1 bf16 transposed [E][H][F]

__device__ __forceinline__ unsigned short f2bf(float f) {
  unsigned int u = __float_as_uint(f);
  u += 0x7fffu + ((u >> 16) & 1u);          // round-to-nearest-even
  return (unsigned short)(u >> 16);
}
__device__ __forceinline__ float gelu_exact(float x) {
  return 0.5f * x * (1.0f + erff(x * 0.70710678118654752440f));
}

// LDS byte offset of a __shared__ pointer: flat shared aperture keeps offset in addr[31:0]
__device__ __forceinline__ unsigned lds_off(const void* p) {
  return (unsigned)(uintptr_t)p;
}
// CDNA5 async copy: 16B global -> LDS, per-lane, tracked by ASYNCcnt (ISA ch10 async ops)
__device__ __forceinline__ void async_lds_b128(unsigned ldsaddr, unsigned voff, unsigned long long saddr) {
  asm volatile("global_load_async_to_lds_b128 %0, %1, %2"
               :: "v"(ldsaddr), "v"(voff), "s"(saddr) : "memory");
}
__device__ __forceinline__ void wait_async0() {
  asm volatile("s_wait_asynccnt 0" ::: "memory");
}

// ---------------- 0: init ----------------
__global__ __launch_bounds__(256) void moe_init(float* __restrict__ out,
                                                int* __restrict__ ctrl,
                                                int* __restrict__ tok) {
  const int i = blockIdx.x * 256 + threadIdx.x;
  if (i < kT * kH)   out[i] = 0.0f;
  if (i < kSlotMax)  tok[i] = -1;
  if (i < 32)        ctrl[i] = 0;
}

// ---------------- 1: fp32 -> bf16 activations ----------------
__global__ __launch_bounds__(256) void moe_cvt_x(const float* __restrict__ x,
                                                 unsigned short* __restrict__ xb) {
  const int i = blockIdx.x * 256 + threadIdx.x;
  if (i < kT * kH) xb[i] = f2bf(x[i]);
}

// ---------------- 1b: weight transpose+convert: src[e][K][N] fp32 -> dst[e][N][K] bf16 ----------------
__global__ __launch_bounds__(256) void moe_cvt_w(const float* __restrict__ src,
                                                 unsigned short* __restrict__ dst,
                                                 int K, int N) {
  const int e  = blockIdx.z;
  const int k0 = blockIdx.x * 64;
  const int n0 = blockIdx.y * 64;
  __shared__ unsigned short t[64 * 72];     // [n][k], row stride 72 (144B, 16B aligned)
  const int tid = threadIdx.x;
  const float* s = src + (size_t)e * K * N;
#pragma unroll
  for (int i = 0; i < 16; ++i) {            // coalesced read along n
    const int idx = tid + 256 * i;
    const int kk = idx >> 6, nn = idx & 63;
    t[nn * 72 + kk] = f2bf(s[(size_t)(k0 + kk) * N + n0 + nn]);
  }
  __syncthreads();
  unsigned short* d = dst + (size_t)e * N * K;
#pragma unroll
  for (int i = 0; i < 2; ++i) {             // coalesced write along k (16B per thread)
    const int w = tid + 256 * i;
    const int nn = w >> 3, kseg = w & 7;
    *(Q4*)(&d[(size_t)(n0 + nn) * K + k0 + kseg * 8]) = *(const Q4*)(&t[nn * 72 + kseg * 8]);
  }
}

// ---------------- 2: router (one wave32 per token) ----------------
__global__ __launch_bounds__(256) void moe_router(const float* __restrict__ x,
                                                  const float* __restrict__ gw,
                                                  float* __restrict__ logits_out,
                                                  int* __restrict__ ctrl,
                                                  int* __restrict__ sel_e,
                                                  float* __restrict__ sel_w) {
  const int wid  = threadIdx.x >> 5;
  const int lane = threadIdx.x & 31;
  const int t = blockIdx.x * 8 + wid;
  const float* xr = x + (size_t)t * kH;

  float acc[kE];
#pragma unroll
  for (int e = 0; e < kE; ++e) acc[e] = 0.0f;

  for (int h = lane; h < kH; h += 32) {
    const float xv = xr[h];
    const float* g = gw + (size_t)h * kE;
#pragma unroll
    for (int e = 0; e < kE; ++e) acc[e] += xv * g[e];
  }
#pragma unroll
  for (int e = 0; e < kE; ++e)
    for (int off = 16; off > 0; off >>= 1)
      acc[e] += __shfl_xor(acc[e], off, 32);

  if (lane == 0) {
    float mx = acc[0];
#pragma unroll
    for (int e = 1; e < kE; ++e) mx = fmaxf(mx, acc[e]);
    float p[kE]; float s = 0.0f;
#pragma unroll
    for (int e = 0; e < kE; ++e) { p[e] = __expf(acc[e] - mx); s += p[e]; }
    const float inv = 1.0f / s;
#pragma unroll
    for (int e = 0; e < kE; ++e) p[e] *= inv;

    int e0 = 0; float p0 = p[0];
#pragma unroll
    for (int e = 1; e < kE; ++e) if (p[e] > p0) { p0 = p[e]; e0 = e; }
    int e1 = -1; float p1 = -1.0f;
#pragma unroll
    for (int e = 0; e < kE; ++e) if (e != e0 && p[e] > p1) { p1 = p[e]; e1 = e; }

#pragma unroll
    for (int e = 0; e < kE; ++e) logits_out[(size_t)t * kE + e] = acc[e];
    sel_e[t * 2 + 0] = e0; sel_w[t * 2 + 0] = p0;
    sel_e[t * 2 + 1] = e1; sel_w[t * 2 + 1] = p1;
    atomicAdd(&ctrl[e0], 1);
    atomicAdd(&ctrl[e1], 1);
  }
}

// ---------------- 3: padded prefix offsets ----------------
__global__ void moe_offsets(int* __restrict__ ctrl) {
  if (threadIdx.x == 0) {
    int off = 0;
    ctrl[16] = 0;
#pragma unroll
    for (int e = 0; e < kE; ++e) {
      off += (ctrl[e] + MT - 1) & ~(MT - 1);
      ctrl[17 + e] = off;
    }
  }
}

// ---------------- 4: token -> slot scatter ----------------
__global__ __launch_bounds__(256) void moe_scatter(const int* __restrict__ sel_e,
                                                   const float* __restrict__ sel_w,
                                                   int* __restrict__ ctrl,
                                                   int* __restrict__ tok,
                                                   float* __restrict__ swt) {
  const int t = blockIdx.x * 256 + threadIdx.x;
  if (t >= kT) return;
#pragma unroll
  for (int k = 0; k < 2; ++k) {
    const int e = sel_e[t * 2 + k];
    const int pos = atomicAdd(&ctrl[8 + e], 1);
    const int s = ctrl[16 + e] + pos;
    tok[s] = t;
    swt[s] = sel_w[t * 2 + k];
  }
}

// ---------------- 5: up-projection: h = gelu(x@Wg) * (x@Wv), double-buffered async ----------------
__global__ __launch_bounds__(256) void moe_ffn_up(const unsigned short* __restrict__ xb,
                                                  const unsigned short* __restrict__ wgT,
                                                  const unsigned short* __restrict__ wvT,
                                                  const int* __restrict__ ctrl,
                                                  const int* __restrict__ tok,
                                                  unsigned short* __restrict__ hws) {
  const int* offsets = ctrl + 16;
  const int sBase = blockIdx.x * MT;
  if (sBase >= offsets[8]) return;
  int e = 0;
  while (sBase >= offsets[e + 1]) ++e;
  const int nBase = blockIdx.y * NT;

  __shared__ unsigned short lA[2][MT * KTP];   // [m][k]
  __shared__ unsigned short lG[2][NT * KTP];   // [n][k]
  __shared__ unsigned short lV[2][NT * KTP];

  const int tid  = threadIdx.x;
  const int wid  = tid >> 5, lane = tid & 31;
  const int msub = wid & 3,  nsub = wid >> 2;
  const int half = lane >> 4, ml  = lane & 15;

  v8f aG0 = {}, aG1 = {}, aV0 = {}, aV1 = {};

  // tile-copy assignments: 256 threads x 16B = one 64x32 bf16 tile
  const int r = tid >> 2, seg = tid & 3;
  const int tkn0 = tok[sBase + r];
  const int tkn = tkn0 < 0 ? 0 : tkn0;             // pad rows read token 0 (never stored to out)
  const unsigned ldsA[2] = { lds_off(&lA[0][r * KTP + seg * 8]), lds_off(&lA[1][r * KTP + seg * 8]) };
  const unsigned ldsG[2] = { lds_off(&lG[0][r * KTP + seg * 8]), lds_off(&lG[1][r * KTP + seg * 8]) };
  const unsigned ldsV[2] = { lds_off(&lV[0][r * KTP + seg * 8]), lds_off(&lV[1][r * KTP + seg * 8]) };
  const unsigned aRowByte = (unsigned)((size_t)tkn * kH * 2 + seg * 16);
  const unsigned wRowByte = (unsigned)((((size_t)e * kF + nBase + r) * kH + seg * 8) * 2);
  const unsigned long long xbA = (unsigned long long)xb;
  const unsigned long long wgA = (unsigned long long)wgT;
  const unsigned long long wvA = (unsigned long long)wvT;

  // prologue: prefetch first K-slab into buffer 0
  async_lds_b128(ldsA[0], aRowByte, xbA);
  async_lds_b128(ldsG[0], wRowByte, wgA);
  async_lds_b128(ldsV[0], wRowByte, wvA);

  constexpr int STEPS = kH / KT;                   // 32
  for (int it = 0; it < STEPS; ++it) {
    const int cur = it & 1;
    wait_async0();                                 // my in-flight loads (into cur) done
    __syncthreads();                               // everyone's loads done; old reads of nxt done
    if (it + 1 < STEPS) {                          // prefetch next slab into other buffer
      const unsigned kb = (unsigned)(it + 1) * (KT * 2);
      async_lds_b128(ldsA[cur ^ 1], aRowByte + kb, xbA);
      async_lds_b128(ldsG[cur ^ 1], wRowByte + kb, wgA);
      async_lds_b128(ldsV[cur ^ 1], wRowByte + kb, wvA);
    }

    Frag fa, fg0, fg1, fv0, fv1;
    const unsigned short* ar = &lA[cur][(msub * 16 + ml) * KTP];
    fa.q[0] = *(const Q4*)(ar + half * 16);        // K 0..7 / 16..23
    fa.q[1] = *(const Q4*)(ar + half * 16 + 8);    // K 8..15 / 24..31
    const int nb0 = (nsub * 32 + ml) * KTP + half * 16;
    const int nb1 = nb0 + 16 * KTP;
    fg0.q[0] = *(const Q4*)(&lG[cur][nb0]); fg0.q[1] = *(const Q4*)(&lG[cur][nb0 + 8]);
    fg1.q[0] = *(const Q4*)(&lG[cur][nb1]); fg1.q[1] = *(const Q4*)(&lG[cur][nb1 + 8]);
    fv0.q[0] = *(const Q4*)(&lV[cur][nb0]); fv0.q[1] = *(const Q4*)(&lV[cur][nb0 + 8]);
    fv1.q[0] = *(const Q4*)(&lV[cur][nb1]); fv1.q[1] = *(const Q4*)(&lV[cur][nb1 + 8]);

    aG0 = __builtin_amdgcn_wmma_f32_16x16x32_bf16(false, fa.v, false, fg0.v, (short)0, aG0, false, false);
    aG1 = __builtin_amdgcn_wmma_f32_16x16x32_bf16(false, fa.v, false, fg1.v, (short)0, aG1, false, false);
    aV0 = __builtin_amdgcn_wmma_f32_16x16x32_bf16(false, fa.v, false, fv0.v, (short)0, aV0, false, false);
    aV1 = __builtin_amdgcn_wmma_f32_16x16x32_bf16(false, fa.v, false, fv1.v, (short)0, aV1, false, false);
  }

#pragma unroll
  for (int j = 0; j < 2; ++j) {
    const v8f g = j ? aG1 : aG0;
    const v8f v = j ? aV1 : aV0;
    const size_t col = (size_t)nBase + nsub * 32 + j * 16 + ml;
#pragma unroll
    for (int i = 0; i < 8; ++i) {                  // D layout: vgpr i -> row i + 8*half
      const int s = sBase + msub * 16 + i + 8 * half;
      hws[(size_t)s * kF + col] = f2bf(gelu_exact(g[i]) * v[i]);
    }
  }
}

// ---------------- 6: down-projection + deterministic scaled atomic combine ----------------
__global__ __launch_bounds__(256) void moe_ffn_down(const unsigned short* __restrict__ hws,
                                                    const unsigned short* __restrict__ w1T,
                                                    const int* __restrict__ ctrl,
                                                    const int* __restrict__ tok,
                                                    const float* __restrict__ swt,
                                                    float* __restrict__ out) {
  const int* offsets = ctrl + 16;
  const int sBase = blockIdx.x * MT;
  if (sBase >= offsets[8]) return;
  int e = 0;
  while (sBase >= offsets[e + 1]) ++e;
  const int nBase = blockIdx.y * NT;

  __shared__ unsigned short lA[2][MT * KTP];
  __shared__ unsigned short lW[2][NT * KTP];

  const int tid  = threadIdx.x;
  const int wid  = tid >> 5, lane = tid & 31;
  const int msub = wid & 3,  nsub = wid >> 2;
  const int half = lane >> 4, ml  = lane & 15;

  v8f a0 = {}, a1 = {};

  const int r = tid >> 2, seg = tid & 3;
  const unsigned ldsA[2] = { lds_off(&lA[0][r * KTP + seg * 8]), lds_off(&lA[1][r * KTP + seg * 8]) };
  const unsigned ldsW[2] = { lds_off(&lW[0][r * KTP + seg * 8]), lds_off(&lW[1][r * KTP + seg * 8]) };
  const unsigned hRowByte = (unsigned)(((size_t)(sBase + r) * kF + seg * 8) * 2);
  const unsigned wRowByte = (unsigned)((((size_t)e * kH + nBase + r) * kF + seg * 8) * 2);
  const unsigned long long hA  = (unsigned long long)hws;
  const unsigned long long w1A = (unsigned long long)w1T;

  async_lds_b128(ldsA[0], hRowByte, hA);
  async_lds_b128(ldsW[0], wRowByte, w1A);

  constexpr int STEPS = kF / KT;                   // 128
  for (int it = 0; it < STEPS; ++it) {
    const int cur = it & 1;
    wait_async0();
    __syncthreads();
    if (it + 1 < STEPS) {
      const unsigned kb = (unsigned)(it + 1) * (KT * 2);
      async_lds_b128(ldsA[cur ^ 1], hRowByte + kb, hA);
      async_lds_b128(ldsW[cur ^ 1], wRowByte + kb, w1A);
    }

    Frag fa, fw0, fw1;
    const unsigned short* ar = &lA[cur][(msub * 16 + ml) * KTP];
    fa.q[0] = *(const Q4*)(ar + half * 16);
    fa.q[1] = *(const Q4*)(ar + half * 16 + 8);
    const int nb0 = (nsub * 32 + ml) * KTP + half * 16;
    const int nb1 = nb0 + 16 * KTP;
    fw0.q[0] = *(const Q4*)(&lW[cur][nb0]); fw0.q[1] = *(const Q4*)(&lW[cur][nb0 + 8]);
    fw1.q[0] = *(const Q4*)(&lW[cur][nb1]); fw1.q[1] = *(const Q4*)(&lW[cur][nb1 + 8]);

    a0 = __builtin_amdgcn_wmma_f32_16x16x32_bf16(false, fa.v, false, fw0.v, (short)0, a0, false, false);
    a1 = __builtin_amdgcn_wmma_f32_16x16x32_bf16(false, fa.v, false, fw1.v, (short)0, a1, false, false);
  }

#pragma unroll
  for (int j = 0; j < 2; ++j) {
    const v8f a = j ? a1 : a0;
    const int col = nBase + nsub * 32 + j * 16 + ml;
#pragma unroll
    for (int i = 0; i < 8; ++i) {
      const int s = sBase + msub * 16 + i + 8 * half;
      const int t = tok[s];
      if (t >= 0)
        unsafeAtomicAdd(&out[(size_t)t * kH + col], a[i] * swt[s]);  // 2 commutative adds/elem
    }
  }
}

// ---------------- host launch ----------------
extern "C" void kernel_launch(void* const* d_in, const int* in_sizes, int n_in,
                              void* d_out, int out_size, void* d_ws, size_t ws_size,
                              hipStream_t stream) {
  (void)in_sizes; (void)n_in; (void)out_size; (void)ws_size;
  const float* x  = (const float*)d_in[0];   // [B,S,H]
  const float* gw = (const float*)d_in[1];   // [H,E]
  const float* wg = (const float*)d_in[2];   // [E,H,F]
  const float* wv = (const float*)d_in[3];   // [E,H,F]
  const float* w1 = (const float*)d_in[4];   // [E,F,H]

  float* out    = (float*)d_out;                       // [T*H]
  float* logits = out + (size_t)kT * kH;               // [T*E]

  char* ws = (char*)d_ws;
  int*            ctrl  = (int*)(ws + offCtrl);
  int*            selE  = (int*)(ws + offSelE);
  float*          selW  = (float*)(ws + offSelW);
  int*            tokm  = (int*)(ws + offTok);
  float*          swt   = (float*)(ws + offSwt);
  unsigned short* xb    = (unsigned short*)(ws + offXbf);
  unsigned short* hws   = (unsigned short*)(ws + offHws);
  unsigned short* wgT   = (unsigned short*)(ws + offWgT);
  unsigned short* wvT   = (unsigned short*)(ws + offWvT);
  unsigned short* w1T   = (unsigned short*)(ws + offW1T);

  const int nElem = kT * kH;
  moe_init  <<<(nElem + 255) / 256, 256, 0, stream>>>(out, ctrl, tokm);
  moe_cvt_x <<<(nElem + 255) / 256, 256, 0, stream>>>(x, xb);
  moe_cvt_w <<<dim3(kH / 64, kF / 64, kE), 256, 0, stream>>>(wg, wgT, kH, kF);
  moe_cvt_w <<<dim3(kH / 64, kF / 64, kE), 256, 0, stream>>>(wv, wvT, kH, kF);
  moe_cvt_w <<<dim3(kF / 64, kH / 64, kE), 256, 0, stream>>>(w1, w1T, kF, kH);
  moe_router<<<kT / 8, 256, 0, stream>>>(x, gw, logits, ctrl, selE, selW);
  moe_offsets<<<1, 32, 0, stream>>>(ctrl);
  moe_scatter<<<kT / 256, 256, 0, stream>>>(selE, selW, ctrl, tokm, swt);
  moe_ffn_up  <<<dim3(kBlkM, kF / NT), 256, 0, stream>>>(xb, wgT, wvT, ctrl, tokm, hws);
  moe_ffn_down<<<dim3(kBlkM, kH / NT), 256, 0, stream>>>(hws, w1T, ctrl, tokm, swt, out);
}
